// LpWindowAttention_32615981645874
// MI455X (gfx1250) — matrix operand order repaired and checked by reading, not compiled, not run on hardware
//
#include <hip/hip_runtime.h>
#include <math.h>

typedef __bf16 bf16_t;
typedef __attribute__((ext_vector_type(16))) __bf16 v16bf;
typedef __attribute__((ext_vector_type(8)))  float  v8f;
typedef __attribute__((address_space(3))) float lds_float;

#define NH 4
#define HD 32
#define CD 128
#define NT 49
#define NP 64
#define NTHREADS 256

// workspace byte offsets
#define WS_QKVW  0                       // 384*128 bf16 = 98304 B
#define WS_PROJW 98304                   // 128*128 bf16 = 32768 B
#define WS_RPB   131072                  // 4*49*49 f32  = 38416 B

// CDNA5 WMMA 16x16x32 bf16 fragment index helpers (ISA 7.12.2 layouts)
__device__ __forceinline__ int a_kidx(int i, int h, int lane) {
  // A 16x32: lanes 0-15 -> K {0..7, 16..23}; lanes 16-31 -> +8
  return ((i < 4) ? (2 * i) : (16 + 2 * (i - 4))) + ((lane >= 16) ? 8 : 0) + h;
}
__device__ __forceinline__ int b_kidx(int i, int h, int lane) {
  // B 32x16: lanes 0-15 -> K {0..7, 16..23}; lanes 16-31 -> +8
  return ((i < 4) ? (2 * i) : (16 + 2 * (i - 4))) + ((lane >= 16) ? 8 : 0) + h;
}

__global__ void lpwin_prep_kernel(const float* __restrict__ qkv_w,
                                  const float* __restrict__ proj_w,
                                  const float* __restrict__ bias_table,
                                  const int*   __restrict__ rel_idx,
                                  bf16_t* __restrict__ qkvw_b,
                                  bf16_t* __restrict__ projw_b,
                                  float*  __restrict__ rpb) {
  const int n_qkv  = 3 * CD * CD;      // 49152
  const int n_proj = CD * CD;          // 16384
  const int n_rpb  = NH * NT * NT;     // 9604
  int total = n_qkv + n_proj + n_rpb;
  for (int i = blockIdx.x * blockDim.x + threadIdx.x; i < total;
       i += gridDim.x * blockDim.x) {
    if (i < n_qkv) {
      qkvw_b[i] = (bf16_t)qkv_w[i];
    } else if (i < n_qkv + n_proj) {
      int j = i - n_qkv;
      projw_b[j] = (bf16_t)proj_w[j];
    } else {
      int j  = i - n_qkv - n_proj;
      int hh = j / (NT * NT);
      int r  = j % (NT * NT);
      rpb[j] = bias_table[rel_idx[r] * NH + hh];
    }
  }
}

__global__ __launch_bounds__(NTHREADS)
void lpwin_attn_kernel(const float* __restrict__ x,
                       const float* __restrict__ mask,
                       const bf16_t* __restrict__ qkvw,
                       const bf16_t* __restrict__ projw,
                       const float* __restrict__ rpb,
                       const float* __restrict__ proj_b,
                       const float* __restrict__ alpha,
                       float* __restrict__ out,
                       int nw) {
  __shared__ bf16_t sQKV[3][NH][NP][33];   // q/k/v, bf16, padded stride
  __shared__ float  sMask[NT * NT];        // shift-window mask for this window
  __shared__ float  sS[NP][66];            // per-head score matrix (f32)
  __shared__ bf16_t sP[NP][64];            // unnormalized exp(S - max), bf16
  __shared__ float  sRinv[NH][NP];         // per-row 1/sum
  __shared__ bf16_t sO[NP][136];           // attn@V output across heads

  const int w    = blockIdx.x;
  const int tid  = threadIdx.x;
  const int wv   = tid >> 5;
  const int lane = tid & 31;
  const int lm   = lane & 15;
  const int lg   = lane >> 4;   // 0 or 1

  // hint: weights are hot, pull toward caches
  __builtin_prefetch(qkvw + lane * 64, 0, 3);

  // ---- stage mask for this window into LDS via the CDNA5 async path ----
  // GLOBAL_LOAD_ASYNC_TO_LDS_B32: copies one dword from global memory into LDS; tracked by ASYNCcnt,
  // fully overlaps the QKV GEMM below; retired with s_wait_asynccnt + barrier.
  {
    const float* mptr = mask + (size_t)(w % nw) * NT * NT;
    for (int i = tid; i < NT * NT; i += NTHREADS) {
      unsigned int ldsa = (unsigned int)(uintptr_t)(lds_float*)&sMask[i];
      const float* ga = mptr + i;
      asm volatile("global_load_async_to_lds_b32 %0, %1, off"
                   :: "v"(ldsa), "v"(ga) : "memory");
    }
  }

  // ---- phase 1: QKV GEMM  (M=64, N=384, K=128) ----
  {
    const int mt = wv >> 1;            // 2 waves per M-tile
    const int am = mt * 16 + lm;       // source row of x (A fragment M index)
    v16bf a[4];
    const float* xr = x + ((size_t)w * NT + am) * CD;
#pragma unroll
    for (int kb = 0; kb < 4; ++kb) {
#pragma unroll
      for (int i = 0; i < 8; ++i)
#pragma unroll
        for (int h2 = 0; h2 < 2; ++h2) {
          int k = kb * 32 + a_kidx(i, h2, lane);
          float v = (am < NT) ? xr[k] : 0.0f;
          a[kb][2 * i + h2] = (bf16_t)v;
        }
    }
    for (int ntl = 0; ntl < 12; ++ntl) {
      int nt = (wv & 1) * 12 + ntl;    // 12 N-tiles per wave
      int o0 = nt * 16;
      v8f c = {};
#pragma unroll
      for (int kb = 0; kb < 4; ++kb) {
        v16bf b;
#pragma unroll
        for (int i = 0; i < 8; ++i)
#pragma unroll
          for (int h2 = 0; h2 < 2; ++h2) {
            int k = kb * 32 + b_kidx(i, h2, lane);
            b[2 * i + h2] = qkvw[(size_t)(o0 + lm) * CD + k];
          }
        c = __builtin_amdgcn_wmma_f32_16x16x32_bf16(false, a[kb], false, b,
                                                    (short)0, c, false, false);
      }
      int which = o0 >> 7;             // 0=q 1=k 2=v
      int head  = (o0 & 127) >> 5;
      int dd0   = o0 & 31;             // 0 or 16
#pragma unroll
      for (int r = 0; r < 8; ++r) {
        int m2 = mt * 16 + r + lg * 8;
        sQKV[which][head][m2][dd0 + lm] = (bf16_t)c[r];
      }
    }
  }
  // retire the async mask copy before the barrier so LDS is coherent
  asm volatile("s_wait_asynccnt 0x0" ::: "memory");
  __syncthreads();

  // ---- phase 2: Lp (p=2) QK-norm, fold alpha into q ----
  for (int rr = tid; rr < 2 * NH * NP; rr += NTHREADS) {
    int which = rr / (NH * NP);
    int rem   = rr % (NH * NP);
    int head  = rem / NP;
    int m     = rem % NP;
    bf16_t* row = &sQKV[which][head][m][0];
    float ss = 0.0f;
#pragma unroll
    for (int d = 0; d < HD; ++d) { float v = (float)row[d]; ss += v * v; }
    float sc = 1.0f / (sqrtf(ss) + 1e-6f);
    if (which == 0) sc *= alpha[head];
#pragma unroll
    for (int d = 0; d < HD; ++d) row[d] = (bf16_t)((float)row[d] * sc);
  }
  __syncthreads();

  // ---- phase 3: per-head attention ----
  for (int hd = 0; hd < NH; ++hd) {
    // S = (alpha*qhat) @ khat^T : 16 tiles, 2 per wave, K=32 (single WMMA)
#pragma unroll
    for (int tt = 0; tt < 2; ++tt) {
      int tile = wv * 2 + tt;
      int mt = tile >> 2, nt = tile & 3;
      v16bf a, b;
      int am = mt * 16 + lm;
#pragma unroll
      for (int i = 0; i < 8; ++i)
#pragma unroll
        for (int h2 = 0; h2 < 2; ++h2) {
          a[2 * i + h2] = sQKV[0][hd][am][a_kidx(i, h2, lane)];
          b[2 * i + h2] = sQKV[1][hd][nt * 16 + lm][b_kidx(i, h2, lane)];
        }
      v8f c = {};
      c = __builtin_amdgcn_wmma_f32_16x16x32_bf16(false, a, false, b,
                                                  (short)0, c, false, false);
#pragma unroll
      for (int r = 0; r < 8; ++r)
        sS[mt * 16 + r + lg * 8][nt * 16 + lm] = c[r];
    }
    __syncthreads();

    // softmax: 4 lanes per row, +bias +mask, keep exp unnormalized in bf16
    {
      int m    = tid >> 2;
      int part = tid & 3;
      float vals[16];
      float mx = -1e30f;
      if (m < NT) {
#pragma unroll
        for (int j = 0; j < 16; ++j) {
          int nn = part * 16 + j;
          if (nn < NT) {
            float v = sS[m][nn] + rpb[(hd * NT + m) * NT + nn] + sMask[m * NT + nn];
            vals[j] = v;
            mx = fmaxf(mx, v);
          } else {
            vals[j] = -1e30f;
          }
        }
      } else {
#pragma unroll
        for (int j = 0; j < 16; ++j) vals[j] = -1e30f;
      }
      mx = fmaxf(mx, __shfl_xor(mx, 1, 4));
      mx = fmaxf(mx, __shfl_xor(mx, 2, 4));
      float sm = 0.0f;
#pragma unroll
      for (int j = 0; j < 16; ++j) {
        int nn = part * 16 + j;
        float e = (m < NT && nn < NT) ? __expf(vals[j] - mx) : 0.0f;
        sP[m][nn] = (bf16_t)e;
        sm += e;
      }
      sm += __shfl_xor(sm, 1, 4);
      sm += __shfl_xor(sm, 2, 4);
      if (part == 0) sRinv[hd][m] = (m < NT) ? (1.0f / sm) : 0.0f;
    }
    __syncthreads();

    // O_h = P @ V : 8 tiles (4 M x 2 N), 1 per wave, K=64 (two WMMAs)
    {
      int mt = wv >> 1, nt2 = wv & 1;
      v8f c = {};
      int am = mt * 16 + lm;
#pragma unroll
      for (int kb = 0; kb < 2; ++kb) {
        v16bf a, b;
#pragma unroll
        for (int i = 0; i < 8; ++i)
#pragma unroll
          for (int h2 = 0; h2 < 2; ++h2) {
            a[2 * i + h2] = sP[am][kb * 32 + a_kidx(i, h2, lane)];
            b[2 * i + h2] = sQKV[2][hd][kb * 32 + b_kidx(i, h2, lane)][nt2 * 16 + lm];
          }
        c = __builtin_amdgcn_wmma_f32_16x16x32_bf16(false, a, false, b,
                                                    (short)0, c, false, false);
      }
#pragma unroll
      for (int r = 0; r < 8; ++r) {
        int m2 = mt * 16 + r + lg * 8;
        sO[m2][hd * 32 + nt2 * 16 + lm] = (bf16_t)(c[r] * sRinv[hd][m2]);
      }
    }
    __syncthreads();
  }

  // ---- phase 4: output projection (M=64, N=128, K=128) + bias, store ----
  for (int tt = 0; tt < 4; ++tt) {
    int tile = wv * 4 + tt;
    int mt = tile >> 3, nt = tile & 7;
    v8f c = {};
    int am = mt * 16 + lm;
#pragma unroll
    for (int kb = 0; kb < 4; ++kb) {
      v16bf a, b;
#pragma unroll
      for (int i = 0; i < 8; ++i)
#pragma unroll
        for (int h2 = 0; h2 < 2; ++h2) {
          a[2 * i + h2] = sO[am][kb * 32 + a_kidx(i, h2, lane)];
          b[2 * i + h2] = projw[(size_t)(nt * 16 + lm) * CD + kb * 32 + b_kidx(i, h2, lane)];
        }
      c = __builtin_amdgcn_wmma_f32_16x16x32_bf16(false, a, false, b,
                                                  (short)0, c, false, false);
    }
    int col = nt * 16 + lm;
    float pb = proj_b[col];
#pragma unroll
    for (int r = 0; r < 8; ++r) {
      int m2 = mt * 16 + r + lg * 8;
      if (m2 < NT) out[((size_t)w * NT + m2) * CD + col] = c[r] + pb;
    }
  }
}

extern "C" void kernel_launch(void* const* d_in, const int* in_sizes, int n_in,
                              void* d_out, int out_size, void* d_ws, size_t ws_size,
                              hipStream_t stream) {
  const float* x          = (const float*)d_in[0];
  const float* mask       = (const float*)d_in[1];
  const float* qkv_w      = (const float*)d_in[2];
  const float* proj_w     = (const float*)d_in[3];
  const float* proj_b     = (const float*)d_in[4];
  const float* bias_table = (const float*)d_in[5];
  const float* alpha      = (const float*)d_in[6];
  const int*   rel_idx    = (const int*)d_in[7];

  const int b  = in_sizes[0] / (NT * CD);        // 8192 windows
  const int nw = in_sizes[1] / (NT * NT);        // 4096 mask windows

  bf16_t* qkvw_b  = (bf16_t*)((char*)d_ws + WS_QKVW);
  bf16_t* projw_b = (bf16_t*)((char*)d_ws + WS_PROJW);
  float*  rpb     = (float*)((char*)d_ws + WS_RPB);

  lpwin_prep_kernel<<<128, NTHREADS, 0, stream>>>(qkv_w, proj_w, bias_table,
                                                  rel_idx, qkvw_b, projw_b, rpb);
  lpwin_attn_kernel<<<b, NTHREADS, 0, stream>>>(x, mask, qkvw_b, projw_b, rpb,
                                                proj_b, alpha, (float*)d_out, nw);
}
